// ScaledDotProductAttention_83554293776546
// MI455X (gfx1250) — matrix operand order repaired and checked by reading, not compiled, not run on hardware
//
#include <hip/hip_runtime.h>

typedef _Float16 half_t;
typedef __attribute__((ext_vector_type(16))) _Float16 v16h;
typedef __attribute__((ext_vector_type(8)))  _Float16 v8h;
typedef __attribute__((ext_vector_type(4)))  _Float16 v4h;
typedef __attribute__((ext_vector_type(8)))  float    v8f;
typedef __attribute__((ext_vector_type(4)))  float    v4f;

#define SEQ_N    8192
#define SEQ_M    8192
#define DIM      256
#define BLOCK_M  128      // queries per workgroup (8 waves x 16)
#define BLOCK_N  32       // keys per iteration
#define NBLK     (SEQ_M / BLOCK_N)
#define NTHREADS 256
#define QT_STRIDE 136     // 128 queries + 8 pad (halfs)  -> conflict-free B frags
#define KV_STRIDE 264     // 256 dims   + 8 pad (halfs)  -> conflict-free A/B frags

// fold 1/sqrt(256) and log2(e) (softmax in base 2) into Q
#define QSCALE (0.0625f * 1.44269504088896340736f)

__global__ __launch_bounds__(NTHREADS) void cvt_f32_to_f16(
    const float* __restrict__ src, half_t* __restrict__ dst) {
  size_t i = ((size_t)blockIdx.x * NTHREADS + threadIdx.x) * 4;
  v4f f = *(const v4f*)(src + i);
  v4h h;
  h.x = (half_t)f.x; h.y = (half_t)f.y; h.z = (half_t)f.z; h.w = (half_t)f.w;
  *(v4h*)(dst + i) = h;
}

// ---- async DMA staging (f16 path): global -> LDS, tracked by ASYNCcnt ----
__device__ __forceinline__ void stage_kv_async(
    unsigned ksBase, unsigned vsBase,
    const half_t* __restrict__ Kh, const half_t* __restrict__ Vh,
    int kb, int tid) {
  #pragma unroll
  for (int j = 0; j < 4; ++j) {
    int i8   = tid + NTHREADS * j;      // b128 chunk index within 32x256 halfs
    int elem = i8 * 8;
    int row  = elem >> 8;
    int col  = elem & 255;
    unsigned ldsK = ksBase + (unsigned)((row * KV_STRIDE + col) * 2);
    unsigned ldsV = vsBase + (unsigned)((row * KV_STRIDE + col) * 2);
    unsigned gof  = (unsigned)((((kb + row) * DIM) + col) * 2);
    asm volatile("global_load_async_to_lds_b128 %0, %1, %2"
                 :: "v"(ldsK), "v"(gof), "s"(Kh) : "memory");
    asm volatile("global_load_async_to_lds_b128 %0, %1, %2"
                 :: "v"(ldsV), "v"(gof), "s"(Vh) : "memory");
  }
}

// ---- synchronous staging with on-the-fly cvt (fp32 fallback path) ----
__device__ __forceinline__ void stage_kv_sync(
    half_t* __restrict__ ks, half_t* __restrict__ vs,
    const float* __restrict__ K, const float* __restrict__ V,
    int kb, int tid) {
  #pragma unroll
  for (int j = 0; j < 8; ++j) {
    int i4   = tid + NTHREADS * j;      // float4 idx in 32x256
    int elem = i4 * 4;
    int row  = elem >> 8;
    int col  = elem & 255;
    v4f fk = *(const v4f*)(K + (size_t)(kb + row) * DIM + col);
    v4f fv = *(const v4f*)(V + (size_t)(kb + row) * DIM + col);
    v4h hk, hv;
    hk.x = (half_t)fk.x; hk.y = (half_t)fk.y; hk.z = (half_t)fk.z; hk.w = (half_t)fk.w;
    hv.x = (half_t)fv.x; hv.y = (half_t)fv.y; hv.z = (half_t)fv.z; hv.w = (half_t)fv.w;
    *(v4h*)&ks[row * KV_STRIDE + col] = hk;
    *(v4h*)&vs[row * KV_STRIDE + col] = hv;
  }
}

template <typename TIn>
__global__ __launch_bounds__(NTHREADS) void fa_fwd(
    const float* __restrict__ Q,
    const TIn*  __restrict__ K,
    const TIn*  __restrict__ V,
    float* __restrict__ out) {

  constexpr bool ASYNC = (sizeof(TIn) == 2);

  __shared__ half_t Qt[DIM * QT_STRIDE];          // Q^T, scaled: [dim][query]
  __shared__ half_t Ks[2][BLOCK_N * KV_STRIDE];   // double-buffered [key][dim]
  __shared__ half_t Vs[2][BLOCK_N * KV_STRIDE];

  const int tid  = threadIdx.x;
  const int lane = tid & 31;
  const int wq   = tid >> 5;     // wave id = local query tile
  const int hh   = lane >> 4;    // half-wave
  const int ln   = lane & 15;
  const int q0   = blockIdx.x * BLOCK_M;

  // ---- stage Q^T (once), fold scale ----
  for (int j = 0; j < 32; ++j) {
    int i4   = tid + NTHREADS * j;      // float4 index in 128x256 tile
    int elem = i4 * 4;
    int row  = elem >> 8;               // local query 0..127
    int col  = elem & 255;              // dim
    v4f f = *(const v4f*)(Q + (size_t)(q0 + row) * DIM + col);
    Qt[(col + 0) * QT_STRIDE + row] = (half_t)(f.x * QSCALE);
    Qt[(col + 1) * QT_STRIDE + row] = (half_t)(f.y * QSCALE);
    Qt[(col + 2) * QT_STRIDE + row] = (half_t)(f.z * QSCALE);
    Qt[(col + 3) * QT_STRIDE + row] = (half_t)(f.w * QSCALE);
  }

  v8f O[16];                       // 16 output tiles: rows=queries(r+8h), col=ln
  #pragma unroll
  for (int t = 0; t < 16; ++t)
    #pragma unroll
    for (int r = 0; r < 8; ++r) O[t][r] = 0.0f;

  float m_i = -1e30f;              // running row max (base-2 domain), per query=ln
  float l_i = 0.0f;                // running row sum

  // ---- prologue: stage block 0 into buffer 0 ----
  if constexpr (ASYNC) {
    stage_kv_async((unsigned)(uintptr_t)&Ks[0][0], (unsigned)(uintptr_t)&Vs[0][0],
                   (const half_t*)K, (const half_t*)V, 0, tid);
  } else {
    stage_kv_sync(&Ks[0][0], &Vs[0][0], (const float*)K, (const float*)V, 0, tid);
  }

  for (int i = 0; i < NBLK; ++i) {
    const int nb = i & 1;
    if constexpr (ASYNC) {
      asm volatile("s_wait_asynccnt 0x0" ::: "memory");  // buffer nb filled (this wave)
    }
    __syncthreads();   // all waves: fill complete; previous readers of nb^1 done

    // ---- stage NEXT block into the other buffer, overlapped with compute ----
    if (i + 1 < NBLK) {
      if constexpr (ASYNC) {
        stage_kv_async((unsigned)(uintptr_t)&Ks[nb ^ 1][0],
                       (unsigned)(uintptr_t)&Vs[nb ^ 1][0],
                       (const half_t*)K, (const half_t*)V, (i + 1) * BLOCK_N, tid);
      } else {
        stage_kv_sync(&Ks[nb ^ 1][0], &Vs[nb ^ 1][0],
                      (const float*)K, (const float*)V, (i + 1) * BLOCK_N, tid);
      }
    }

    const half_t* __restrict__ ks = &Ks[nb][0];
    const half_t* __restrict__ vs = &Vs[nb][0];

    // ---- S^T = K_blk · Q^T : two 16(key)x16(query) tiles ----
    v8f st0, st1;
    #pragma unroll
    for (int r = 0; r < 8; ++r) { st0[r] = 0.0f; st1[r] = 0.0f; }
    #pragma unroll
    for (int s = 0; s < 8; ++s) {      // dim steps of 32
      v16h bq;                         // B = Q^T[32 dims][16 queries]: lane&15=query, halfs=dims {8h+0..7, 8h+16..23}
      #pragma unroll
      for (int e = 0; e < 16; ++e) {
        int kk = (e < 8) ? (hh * 8 + e) : (16 + hh * 8 + (e - 8));
        bq[e] = Qt[(s * 32 + kk) * QT_STRIDE + wq * 16 + ln];
      }
      v16h a0, a1;                     // A = K rows: lane&15=key, halfs=dims {8h+0..7, 8h+16..23}
      const half_t* kp0 = &ks[ln * KV_STRIDE + s * 32 + hh * 8];
      ((v8h*)&a0)[0] = *(const v8h*)kp0;
      ((v8h*)&a0)[1] = *(const v8h*)(kp0 + 16);
      const half_t* kp1 = kp0 + 16 * KV_STRIDE;
      ((v8h*)&a1)[0] = *(const v8h*)kp1;
      ((v8h*)&a1)[1] = *(const v8h*)(kp1 + 16);
      st0 = __builtin_amdgcn_wmma_f32_16x16x32_f16(false, a0, false, bq, (short)0, st0, false, false);
      st1 = __builtin_amdgcn_wmma_f32_16x16x32_f16(false, a1, false, bq, (short)0, st1, false, false);
    }

    // ---- online softmax; lane holds query=ln, keys kb + hh*8 + r (+16 for st1) ----
    float cmax = -1e30f;
    #pragma unroll
    for (int r = 0; r < 8; ++r) cmax = fmaxf(cmax, fmaxf(st0[r], st1[r]));
    cmax = fmaxf(cmax, __shfl_xor(cmax, 16, 32));       // other half covers other 16 keys
    const float mnew  = fmaxf(m_i, cmax);
    const float alpha = exp2f(m_i - mnew);
    float csum = 0.0f;
    #pragma unroll
    for (int r = 0; r < 8; ++r) {
      float p0 = exp2f(st0[r] - mnew);
      float p1 = exp2f(st1[r] - mnew);
      st0[r] = p0; st1[r] = p1;
      csum += p0 + p1;
    }
    csum += __shfl_xor(csum, 16, 32);
    l_i = l_i * alpha + csum;
    m_i = mnew;

    // ---- P as WMMA A operand: pure in-register relayout (cvt only) ----
    v16h pa;
    #pragma unroll
    for (int r = 0; r < 8; ++r) {
      pa[r]     = (half_t)st0[r];
      pa[r + 8] = (half_t)st1[r];
    }

    // ---- rescale O accumulators by alpha of their rows ----
    float fac[8];
    #pragma unroll
    for (int r = 0; r < 8; ++r) fac[r] = __shfl(alpha, hh * 8 + r, 32);
    #pragma unroll
    for (int t = 0; t < 16; ++t)
      #pragma unroll
      for (int r = 0; r < 8; ++r) O[t][r] *= fac[r];

    // ---- O += P · V_blk ; B = V: lane&15=vcol, halfs=keys {8h+0..7, 8h+16..23} ----
    #pragma unroll
    for (int t = 0; t < 16; ++t) {
      v16h bv;
      #pragma unroll
      for (int e = 0; e < 16; ++e) {
        int kk = (e < 8) ? (hh * 8 + e) : (16 + hh * 8 + (e - 8));
        bv[e] = vs[kk * KV_STRIDE + t * 16 + ln];
      }
      O[t] = __builtin_amdgcn_wmma_f32_16x16x32_f16(false, pa, false, bv, (short)0, O[t], false, false);
    }
  }

  // ---- finalize: divide by row sums, store fp32 ----
  const float inv = 1.0f / l_i;
  float fac[8];
  #pragma unroll
  for (int r = 0; r < 8; ++r) fac[r] = __shfl(inv, hh * 8 + r, 32);
  #pragma unroll
  for (int t = 0; t < 16; ++t) {
    #pragma unroll
    for (int r = 0; r < 8; ++r) {
      size_t row = (size_t)q0 + wq * 16 + hh * 8 + r;
      out[row * DIM + t * 16 + ln] = O[t][r] * fac[r];
    }
  }
}

extern "C" void kernel_launch(void* const* d_in, const int* in_sizes, int n_in,
                              void* d_out, int out_size, void* d_ws, size_t ws_size,
                              hipStream_t stream) {
  (void)in_sizes; (void)n_in; (void)out_size;
  const float* Q = (const float*)d_in[0];
  const float* K = (const float*)d_in[1];
  const float* V = (const float*)d_in[2];
  float* out = (float*)d_out;

  const size_t elems = (size_t)SEQ_M * DIM;              // 2M elements each
  const size_t need  = 2 * elems * sizeof(half_t);       // K+V in f16

  if (ws_size >= need) {
    half_t* Kh = (half_t*)d_ws;
    half_t* Vh = Kh + elems;
    const int cvt_blocks = (int)(elems / (4 * NTHREADS)); // 2048
    cvt_f32_to_f16<<<cvt_blocks, NTHREADS, 0, stream>>>(K, Kh);
    cvt_f32_to_f16<<<cvt_blocks, NTHREADS, 0, stream>>>(V, Vh);
    fa_fwd<half_t><<<SEQ_N / BLOCK_M, NTHREADS, 0, stream>>>(Q, Kh, Vh, out);
  } else {
    fa_fwd<float><<<SEQ_N / BLOCK_M, NTHREADS, 0, stream>>>(Q, K, V, out);
  }
}
